// MambaLayer_10041633538693
// MI455X (gfx1250) — matrix operand order repaired and checked by reading, not compiled, not run on hardware
//
#include <hip/hip_runtime.h>
#include <hip/hip_bf16.h>
#include <math.h>

// ---------------- problem constants (from reference) ----------------
#define BQ       8
#define DMODEL   256
#define LIMG     4096          // 16*16*16
#define DINNER   512
#define DSTATE   16
#define DTRANK   16
#define DCONV    4
#define MROWS    (BQ * LIMG)   // 32768
#define TCHUNK   32            // scan timesteps staged per barrier

// ---------------- vector types for WMMA ----------------
typedef __attribute__((ext_vector_type(16))) __bf16 v16bf;
typedef __attribute__((ext_vector_type(8)))  __bf16 bf16x8;
typedef __attribute__((ext_vector_type(8)))  float  v8f;

static __device__ __forceinline__ v16bf make16(bf16x8 lo, bf16x8 hi) {
    v16bf r;
    #pragma unroll
    for (int i = 0; i < 8; ++i) { r[i] = lo[i]; r[8 + i] = hi[i]; }
    return r;
}

// =====================================================================
// Weight prep: fp32 (N,K) row-major  ->  bf16 (K,N) row-major
// =====================================================================
__global__ void w_to_bf16_t(const float* __restrict__ W, __bf16* __restrict__ Wt,
                            int N, int K) {
    int i = blockIdx.x * 256 + threadIdx.x;
    if (i >= N * K) return;
    int n = i / K, k = i % K;
    Wt[(size_t)k * N + n] = (__bf16)W[i];
}

__global__ void neg_exp_k(const float* __restrict__ a, float* __restrict__ o, int n) {
    int i = blockIdx.x * 256 + threadIdx.x;
    if (i < n) o[i] = -__expf(a[i]);
}

// =====================================================================
// LayerNorm over 256 channels. mode 1: input in (B, D, L) layout
// (strided gather == fused transpose). mode 0: contiguous (row, D).
// =====================================================================
__global__ __launch_bounds__(256) void ln_kernel(const float* __restrict__ in,
                                                 __bf16* __restrict__ out,
                                                 const float* __restrict__ w,
                                                 const float* __restrict__ bias,
                                                 int mode) {
    __shared__ float red[256];
    int row = blockIdx.x;
    int d   = threadIdx.x;
    float v;
    if (mode == 1) {
        int b = row >> 12, l = row & (LIMG - 1);
        v = in[((size_t)b * DMODEL + d) * LIMG + l];
    } else {
        v = in[(size_t)row * DMODEL + d];
    }
    red[d] = v; __syncthreads();
    for (int s = 128; s > 0; s >>= 1) { if (d < s) red[d] += red[d + s]; __syncthreads(); }
    float mu = red[0] * (1.0f / DMODEL);
    __syncthreads();
    float diff = v - mu;
    red[d] = diff * diff; __syncthreads();
    for (int s = 128; s > 0; s >>= 1) { if (d < s) red[d] += red[d + s]; __syncthreads(); }
    float rstd = rsqrtf(red[0] * (1.0f / DMODEL) + 1e-5f);
    out[(size_t)row * DMODEL + d] = (__bf16)(diff * rstd * w[d] + bias[d]);
}

// =====================================================================
// WMMA GEMM:  C[M, N] = A[M, K](bf16, LDA) * Bt[K, N](bf16), f32 acc.
// Compile-time shapes; one wave computes a 16 x (NT*16) tile with NT
// independent accumulators (A reuse x NT, hazards covered).
// Explicit 1-stage software pipeline: next K-step's A/B fragments are
// loaded BEFORE the current K-step's WMMAs, so the loadcnt wait in
// front of each WMMA group covers loads issued a full iteration ago.
// Block = 8 waves = 128 consecutive M rows.
// EPI 0: store f32 (+ optional bf16 mirror)
// EPI 1: softplus(C + bias) -> f32        (dt_proj -> delta)
// EPI 2: C + bias + residual, store transposed to (B, D, L)
// =====================================================================
template <int K, int LDA, int N, int NT, int EPI>
__global__ __launch_bounds__(256) void wmma_gemm(
    const __bf16* __restrict__ A,
    const __bf16* __restrict__ Bt,
    float* Cf, __bf16* Cb,
    const float* __restrict__ bias,
    const float* __restrict__ resid, float* outT)
{
    const int wave = threadIdx.x >> 5;
    const int lane = threadIdx.x & 31;
    const int mt   = blockIdx.x * 8 + wave;       // 16-row tile index
    const int nt0  = blockIdx.y * NT;             // first 16-col tile index

    const int half  = lane >> 4;                  // 0/1
    const int kbase = half * 8;
    const int mrow  = mt * 16 + (lane & 15);

    // A: ISA 16-bit 16x32 layout -> per lane two contiguous 8-elem chunks
    const __bf16* __restrict__ arow = A + (size_t)mrow * LDA + kbase;
    // B: lane holds column n = lane&15, k = kbase+i and 16+kbase+i (same order as A)
    const __bf16* __restrict__ brow = Bt + (size_t)kbase * N + nt0 * 16 + (lane & 15);

    v8f acc[NT];
    #pragma unroll
    for (int j = 0; j < NT; ++j) acc[j] = (v8f){};

    constexpr int KMAIN = K & ~31;

    if constexpr (KMAIN > 0) {
        // ---- pipeline prologue: fragments for kt = 0 ----
        bf16x8 a0 = *(const bf16x8*)(arow);
        bf16x8 a1 = *(const bf16x8*)(arow + 16);
        bf16x8 b0[NT], b1[NT];
        #pragma unroll
        for (int j = 0; j < NT; ++j) {
            const __bf16* bp = brow + j * 16;
            #pragma unroll
            for (int i = 0; i < 8; ++i) { b0[j][i] = bp[(size_t)i * N]; b1[j][i] = bp[(size_t)(16 + i) * N]; }
        }

        #pragma unroll 4
        for (int kt = 32; kt < KMAIN; kt += 32) {
            // ---- issue NEXT K-step's loads first ----
            bf16x8 na0 = *(const bf16x8*)(arow + kt);
            bf16x8 na1 = *(const bf16x8*)(arow + kt + 16);
            bf16x8 nb0[NT], nb1[NT];
            #pragma unroll
            for (int j = 0; j < NT; ++j) {
                const __bf16* bp = brow + (size_t)kt * N + j * 16;
                #pragma unroll
                for (int i = 0; i < 8; ++i) { nb0[j][i] = bp[(size_t)i * N]; nb1[j][i] = bp[(size_t)(16 + i) * N]; }
            }
            // ---- consume CURRENT fragments ----
            v16bf afrag = make16(a0, a1);
            #pragma unroll
            for (int j = 0; j < NT; ++j)
                acc[j] = __builtin_amdgcn_wmma_f32_16x16x32_bf16(
                             false, afrag, false, make16(b0[j], b1[j]),
                             (short)0, acc[j], false, false);
            // ---- rotate (renamed away by unrolling) ----
            a0 = na0; a1 = na1;
            #pragma unroll
            for (int j = 0; j < NT; ++j) { b0[j] = nb0[j]; b1[j] = nb1[j]; }
        }

        // ---- pipeline epilogue: last K-step ----
        v16bf afrag = make16(a0, a1);
        #pragma unroll
        for (int j = 0; j < NT; ++j)
            acc[j] = __builtin_amdgcn_wmma_f32_16x16x32_bf16(
                         false, afrag, false, make16(b0[j], b1[j]),
                         (short)0, acc[j], false, false);
    }

    if constexpr ((K % 32) != 0) {                // only dt_proj (K=16) takes this
        constexpr int kt = KMAIN;
        bf16x8 a0 = {}, a1 = {};
        if (kt + kbase + 8 <= K)      a0 = *(const bf16x8*)(arow + kt);
        if (kt + 16 + kbase + 8 <= K) a1 = *(const bf16x8*)(arow + kt + 16);
        bf16x8 b0[NT], b1[NT];
        #pragma unroll
        for (int j = 0; j < NT; ++j) { b0[j] = (bf16x8){}; b1[j] = (bf16x8){}; }
        #pragma unroll
        for (int j = 0; j < NT; ++j) {
            const __bf16* bp = brow + (size_t)kt * N + j * 16;
            #pragma unroll
            for (int i = 0; i < 8; ++i) {
                if (kt + kbase + 8 <= K)      b0[j][i] = bp[(size_t)i * N];
                if (kt + 16 + kbase + 8 <= K) b1[j][i] = bp[(size_t)(16 + i) * N];
            }
        }
        v16bf afrag = make16(a0, a1);
        #pragma unroll
        for (int j = 0; j < NT; ++j)
            acc[j] = __builtin_amdgcn_wmma_f32_16x16x32_bf16(
                         false, afrag, false, make16(b0[j], b1[j]),
                         (short)0, acc[j], false, false);
    }

    // ---- epilogue; C layout: lane -> col = lane&15, VGPR r -> row r + 8*half
    #pragma unroll
    for (int j = 0; j < NT; ++j) {
        const int ncol = (nt0 + j) * 16 + (lane & 15);
        const float bv = bias ? bias[ncol] : 0.0f;
        #pragma unroll
        for (int r = 0; r < 8; ++r) {
            int mr = mt * 16 + r + 8 * half;
            float v = acc[j][r] + bv;
            if (EPI == 1) v = (v > 20.0f) ? v : log1pf(__expf(v));
            if (EPI == 2) {
                int bi = mr >> 12, li = mr & (LIMG - 1);
                size_t oi = ((size_t)bi * DMODEL + ncol) * LIMG + li;
                outT[oi] = v + resid[oi];
            } else {
                if (Cf) Cf[(size_t)mr * N + ncol] = v;
                if (Cb) Cb[(size_t)mr * N + ncol] = (__bf16)v;
            }
        }
    }
}

// =====================================================================
// Depthwise causal conv (k=4) + bias + SiLU. Input x_in = xz[:, :512].
// =====================================================================
__global__ __launch_bounds__(256) void conv_silu(const float* __restrict__ xz,
                                                 const float* __restrict__ cw,
                                                 const float* __restrict__ cb,
                                                 float* __restrict__ xc_f,
                                                 __bf16* __restrict__ xc_b) {
    size_t idx = (size_t)blockIdx.x * 256 + threadIdx.x;    // over B*L*DINNER
    int d  = (int)(idx & (DINNER - 1));
    size_t bl = idx >> 9;
    int l = (int)(bl & (LIMG - 1)), b = (int)(bl >> 12);
    float acc = cb[d];
    #pragma unroll
    for (int j = 0; j < DCONV; ++j) {
        int ls = l - (DCONV - 1) + j;
        if (ls >= 0)
            acc += cw[d * DCONV + j] * xz[((size_t)b * LIMG + ls) * (2 * DINNER) + d];
    }
    float s = acc / (1.0f + __expf(-acc));   // silu
    xc_f[idx] = s;
    xc_b[idx] = (__bf16)s;
}

// =====================================================================
// Selective scan fused with D*u skip and SiLU(z) gating.
// One thread per (b,d) channel; h[16] in VGPRs. B/C vectors staged in
// LDS TCHUNK timesteps at a time (coalesced float4 -> ds_store_b128),
// so only 2 barriers per 32 steps; inner loop reads LDS broadcasts.
// grid = 16 blocks (b*2 + half), 256 threads.
// =====================================================================
__global__ __launch_bounds__(256) void scan_gate(const float* __restrict__ delta,
                                                 const float* __restrict__ u,
                                                 const float* __restrict__ xdbl,
                                                 const float* __restrict__ Aneg,
                                                 const float* __restrict__ Dp,
                                                 const float* __restrict__ xz,
                                                 __bf16* __restrict__ y) {
    __shared__ float sBC[TCHUNK][32];             // [step][B:0..15 | C:16..31]
    const int b = blockIdx.x >> 1;
    const int d = ((blockIdx.x & 1) << 8) + threadIdx.x;   // 0..511

    float A[DSTATE], h[DSTATE];
    #pragma unroll
    for (int n = 0; n < DSTATE; ++n) { A[n] = Aneg[d * DSTATE + n]; h[n] = 0.0f; }
    const float Dpar = Dp[d];

    // each thread stages 4 consecutive floats (one float4) per chunk
    const int sidx = (threadIdx.x * 4) >> 5;      // step within chunk
    const int jidx = (threadIdx.x * 4) & 31;      // element within B|C row

    for (int t0 = 0; t0 < LIMG; t0 += TCHUNK) {
        const float4 v4 = *(const float4*)(
            &xdbl[((size_t)b * LIMG + t0 + sidx) * (DTRANK + 2 * DSTATE) + DTRANK + jidx]);
        *(float4*)(&sBC[sidx][jidx]) = v4;
        if (t0 + TCHUNK < LIMG)                   // warm next chunk (global_prefetch)
            __builtin_prefetch(
                &xdbl[((size_t)b * LIMG + t0 + TCHUNK + sidx) * (DTRANK + 2 * DSTATE)], 0, 1);
        __syncthreads();

        for (int s = 0; s < TCHUNK; ++s) {
            const size_t row = (size_t)b * LIMG + t0 + s;
            const float dt = delta[row * DINNER + d];
            const float ut = u[row * DINNER + d];
            const float du = dt * ut;
            float yt = 0.0f;
            #pragma unroll
            for (int n = 0; n < DSTATE; ++n) {
                h[n] = __expf(dt * A[n]) * h[n] + du * sBC[s][n];
                yt  += h[n] * sBC[s][DSTATE + n];
            }
            const float zt = xz[row * (2 * DINNER) + DINNER + d];
            const float g  = zt / (1.0f + __expf(-zt));    // silu(z)
            y[row * DINNER + d] = (__bf16)((yt + ut * Dpar) * g);
        }
        __syncthreads();
    }
}

// =====================================================================
// Host-side orchestration
// =====================================================================
extern "C" void kernel_launch(void* const* d_in, const int* in_sizes, int n_in,
                              void* d_out, int out_size, void* d_ws, size_t ws_size,
                              hipStream_t stream) {
    const float* x         = (const float*)d_in[0];
    const float* norm_w    = (const float*)d_in[1];
    const float* norm_b    = (const float*)d_in[2];
    const float* in_proj_w = (const float*)d_in[3];   // (1024, 256)
    const float* conv_w    = (const float*)d_in[4];   // (512, 1, 4)
    const float* conv_b    = (const float*)d_in[5];
    const float* x_proj_w  = (const float*)d_in[6];   // (48, 512)
    const float* dt_proj_w = (const float*)d_in[7];   // (512, 16)
    const float* dt_proj_b = (const float*)d_in[8];
    const float* A_log     = (const float*)d_in[9];   // (512, 16)
    const float* D_param   = (const float*)d_in[10];
    const float* out_proj_w= (const float*)d_in[11];  // (256, 512)
    const float* proj_w    = (const float*)d_in[12];  // (256, 256)
    const float* proj_b    = (const float*)d_in[13];
    float* outp            = (float*)d_out;

    // ---- workspace carve-up (256B aligned) ----
    char* ws = (char*)d_ws;
    size_t off = 0;
    auto take = [&](size_t bytes) -> char* {
        char* p = ws + off;
        off += (bytes + 255) & ~(size_t)255;
        return p;
    };
    __bf16* xn_bf   = (__bf16*)take((size_t)MROWS * DMODEL * 2);
    float*  xz_f    = (float*) take((size_t)MROWS * 2 * DINNER * 4);
    float*  xc_f    = (float*) take((size_t)MROWS * DINNER * 4);
    __bf16* xc_bf   = (__bf16*)take((size_t)MROWS * DINNER * 2);
    float*  xdbl_f  = (float*) take((size_t)MROWS * 48 * 4);
    __bf16* xdbl_bf = (__bf16*)take((size_t)MROWS * 48 * 2);
    float*  delta_f = (float*) take((size_t)MROWS * DINNER * 4);
    __bf16* y_bf    = (__bf16*)take((size_t)MROWS * DINNER * 2);
    float*  ym_f    = (float*) take((size_t)MROWS * DMODEL * 4);
    __bf16* ym_bf   = (__bf16*)take((size_t)MROWS * DMODEL * 2);
    __bf16* wt_in   = (__bf16*)take((size_t)DMODEL * 2 * DINNER * 2);
    __bf16* wt_xp   = (__bf16*)take((size_t)DINNER * 48 * 2);
    __bf16* wt_dt   = (__bf16*)take((size_t)DTRANK * DINNER * 2);
    __bf16* wt_out  = (__bf16*)take((size_t)DINNER * DMODEL * 2);
    __bf16* wt_pj   = (__bf16*)take((size_t)DMODEL * DMODEL * 2);
    float*  Aneg    = (float*) take((size_t)DINNER * DSTATE * 4);

    // ---- weight prep (tiny; stays hot in 192MB L2 afterwards) ----
    w_to_bf16_t<<<(2*DINNER*DMODEL + 255)/256, 256, 0, stream>>>(in_proj_w,  wt_in,  2*DINNER, DMODEL);
    w_to_bf16_t<<<(48*DINNER       + 255)/256, 256, 0, stream>>>(x_proj_w,   wt_xp,  48,       DINNER);
    w_to_bf16_t<<<(DINNER*DTRANK   + 255)/256, 256, 0, stream>>>(dt_proj_w,  wt_dt,  DINNER,   DTRANK);
    w_to_bf16_t<<<(DMODEL*DINNER   + 255)/256, 256, 0, stream>>>(out_proj_w, wt_out, DMODEL,   DINNER);
    w_to_bf16_t<<<(DMODEL*DMODEL   + 255)/256, 256, 0, stream>>>(proj_w,     wt_pj,  DMODEL,   DMODEL);
    neg_exp_k  <<<(DINNER*DSTATE   + 255)/256, 256, 0, stream>>>(A_log, Aneg, DINNER*DSTATE);

    // ---- 1) LN (+transpose) ----
    ln_kernel<<<MROWS, 256, 0, stream>>>(x, xn_bf, norm_w, norm_b, 1);

    // ---- 2) in_proj: (32768,256)x(256,1024) ----
    wmma_gemm<256, 256, 1024, 4, 0><<<dim3(MROWS/128, 16), 256, 0, stream>>>(
        xn_bf, wt_in, xz_f, nullptr, nullptr, nullptr, nullptr);

    // ---- 3) depthwise conv + SiLU ----
    conv_silu<<<(MROWS/256)*DINNER, 256, 0, stream>>>(xz_f, conv_w, conv_b, xc_f, xc_bf);

    // ---- 4) x_proj: (32768,512)x(512,48) ----
    wmma_gemm<512, 512, 48, 3, 0><<<dim3(MROWS/128, 1), 256, 0, stream>>>(
        xc_bf, wt_xp, xdbl_f, xdbl_bf, nullptr, nullptr, nullptr);

    // ---- 5) dt_proj + softplus: (32768,16)x(16,512), A = x_dbl[:, :16] ----
    wmma_gemm<16, 48, 512, 4, 1><<<dim3(MROWS/128, 8), 256, 0, stream>>>(
        xdbl_bf, wt_dt, delta_f, nullptr, dt_proj_b, nullptr, nullptr);

    // ---- 6) selective scan + D*u + SiLU(z) gate ----
    scan_gate<<<2*BQ, 256, 0, stream>>>(delta_f, xc_f, xdbl_f, Aneg, D_param, xz_f, y_bf);

    // ---- 7) out_proj: (32768,512)x(512,256) ----
    wmma_gemm<512, 512, 256, 4, 0><<<dim3(MROWS/128, 4), 256, 0, stream>>>(
        y_bf, wt_out, ym_f, nullptr, nullptr, nullptr, nullptr);

    // ---- 8) second LN ----
    ln_kernel<<<MROWS, 256, 0, stream>>>(ym_f, ym_bf, norm_w, norm_b, 0);

    // ---- 9) final proj + bias + residual (+ transpose back to (B,D,L)) ----
    wmma_gemm<256, 256, 256, 4, 2><<<dim3(MROWS/128, 4), 256, 0, stream>>>(
        ym_bf, wt_pj, nullptr, nullptr, proj_b, x, outp);

    (void)in_sizes; (void)n_in; (void)out_size; (void)ws_size;
}
